// Attention_41953240547587
// MI455X (gfx1250) — matrix operand order repaired and checked by reading, not compiled, not run on hardware
//
#include <hip/hip_runtime.h>

typedef __attribute__((ext_vector_type(16))) _Float16 v16h;
typedef __attribute__((ext_vector_type(8)))  _Float16 v8h;
typedef __attribute__((ext_vector_type(8)))  float    v8f;

#define HEADS 16
#define HD    64
#define DIMS  1024
#define SEQ   2048
#define BATCH 2
#define MTOT  (BATCH * SEQ)      // 4096 rows (b*s)
#define QK_SCALE 0.125f          // 1/sqrt(64)

// ---------------------------------------------------------------------------
// fp32 -> f16 conversion (grid-stride)
// ---------------------------------------------------------------------------
__global__ void f32_to_f16_kernel(const float* __restrict__ src,
                                  _Float16* __restrict__ dst, int n) {
  int i = blockIdx.x * blockDim.x + threadIdx.x;
  int stride = gridDim.x * blockDim.x;
  for (; i < n; i += stride) dst[i] = (_Float16)src[i];
}

// Assemble an A-fragment (16x32 f16): per lane, halves K+[0..7] and K+[8..15]
// (lane>=16 adds +16 to K) -> two contiguous 16B chunks.
__device__ __forceinline__ v16h load_a_frag(const _Float16* lo,
                                            const _Float16* hi) {
  v8h a0 = *(const v8h*)lo;
  v8h a1 = *(const v8h*)hi;
  v16h a;
#pragma unroll
  for (int i = 0; i < 8; ++i) { a[i] = a0[i]; a[i + 8] = a1[i]; }
  return a;
}

// ---------------------------------------------------------------------------
// Generic f16 GEMM: C[M,N] = A[M,K] * B[N,K]^T  (both K-contiguous, row-major)
// Block = 128 threads = 4 waves. Block tile: 128 rows x 64 cols.
// Wave w computes a 32x64 C tile (2 M-subtiles x 4 N-subtiles, 8 accums):
// each B fragment feeds 2 WMMAs, each A fragment feeds 4 -> 1.5 b128
// loads per WMMA instead of 2.5.
// mode 0: route columns into q (scaled), k, v-transposed f16 buffers (N=3072)
// mode 1: write fp32 out + bias (N=1024)
// ---------------------------------------------------------------------------
__global__ __launch_bounds__(128) void gemm_f16_kernel(
    const _Float16* __restrict__ A, const _Float16* __restrict__ B,
    int K, int N, int mode,
    _Float16* __restrict__ qb, _Float16* __restrict__ kb,
    _Float16* __restrict__ vtb,
    float* __restrict__ outp, const float* __restrict__ bias) {
  const int lane = threadIdx.x & 31;
  const int wave = threadIdx.x >> 5;
  const int mBase = blockIdx.x * 128 + wave * 32;
  const int nBase = blockIdx.y * 64;
  const int n15 = lane & 15;
  const int hiA = (lane >> 4) * 8;    // C row offset
  const int hiB = (lane >> 4) * 16;   // B K-offset

  const _Float16* ap0 = A + (size_t)(mBase + n15) * K + hiB;
  const _Float16* ap1 = ap0 + (size_t)16 * K;
  const _Float16* bp  = B + (size_t)(nBase + n15) * K + hiB;

  v8f zero = {};
  v8f acc[2][4];
#pragma unroll
  for (int mi = 0; mi < 2; ++mi)
#pragma unroll
    for (int j = 0; j < 4; ++j) acc[mi][j] = zero;

  for (int k0 = 0; k0 < K; k0 += 32) {
    // prefetch the next K-chunk of the A streams (lowers to global_prefetch_b8)
    if (k0 + 128 < K) {
      __builtin_prefetch(ap0 + k0 + 128, 0, 1);
      __builtin_prefetch(ap1 + k0 + 128, 0, 1);
    }
    v16h a0 = load_a_frag(ap0 + k0, ap0 + k0 + 8);
    v16h a1 = load_a_frag(ap1 + k0, ap1 + k0 + 8);
#pragma unroll
    for (int j = 0; j < 4; ++j) {
      v16h b = *(const v16h*)(bp + (size_t)(j * 16) * K + k0);
      acc[0][j] = __builtin_amdgcn_wmma_f32_16x16x32_f16(
          false, a0, false, b, (short)0, acc[0][j], false, false);
      acc[1][j] = __builtin_amdgcn_wmma_f32_16x16x32_f16(
          false, a1, false, b, (short)0, acc[1][j], false, false);
    }
  }

#pragma unroll
  for (int mi = 0; mi < 2; ++mi) {
#pragma unroll
    for (int j = 0; j < 4; ++j) {
      int col = nBase + j * 16 + n15;
#pragma unroll
      for (int r = 0; r < 8; ++r) {
        int row = mBase + mi * 16 + r + hiA;
        float v = acc[mi][j][r];
        if (mode == 1) {
          outp[(size_t)row * N + col] = v + bias[col];
        } else {
          int b_ = row / SEQ, s = row - b_ * SEQ;
          if (col < DIMS) {
            int h = col / HD, d = col - h * HD;
            qb[(((size_t)(b_ * HEADS + h)) * SEQ + s) * HD + d] =
                (_Float16)(v * QK_SCALE);
          } else if (col < 2 * DIMS) {
            int c = col - DIMS;
            int h = c / HD, d = c - h * HD;
            kb[(((size_t)(b_ * HEADS + h)) * SEQ + s) * HD + d] = (_Float16)v;
          } else {
            int c = col - 2 * DIMS;
            int h = c / HD, d = c - h * HD;
            vtb[(((size_t)(b_ * HEADS + h)) * HD + d) * SEQ + s] = (_Float16)v;
          }
        }
      }
    }
  }
}

// ---------------------------------------------------------------------------
// Attention core: one block (128 thr = 4 waves) per (b, h, 16-query block).
// Wave w owns keys [w*512, w*512+512).
//   scores (WMMA, d-loop of 2) -> LDS f32 [16][2048]
//   cross-wave softmax stats -> unnormalized exp -> LDS f16 P [16][2048]
//   P * V (WMMA over keys) -> cross-wave reduce -> /rowsum -> f16 out
// LDS: 131072 (scores, reused for out-reduction) + 65536 (P) + stats.
// ---------------------------------------------------------------------------
__global__ __launch_bounds__(128) void attn_kernel(
    const _Float16* __restrict__ qh, const _Float16* __restrict__ kh,
    const _Float16* __restrict__ vth, _Float16* __restrict__ attnh) {
  const int bid = blockIdx.x;
  const int qblk = bid & 127;
  const int h = (bid >> 7) & 15;
  const int b = bid >> 11;

  extern __shared__ char smem[];
  float*    sc  = (float*)smem;                       // [16][2048] f32
  _Float16* p   = (_Float16*)(smem + 131072);         // [16][2048] f16
  float*    red = (float*)(smem + 131072 + 65536);    // [16][16]
  float* rowmax = red + 16 * 16;                      // [16]
  float* rowsum = rowmax + 16;                        // [16]
  float* outred = sc;                                 // reuse: [4][16][64]

  const int lane = threadIdx.x & 31;
  const int wave = threadIdx.x >> 5;
  const int tid = threadIdx.x;
  const int n15 = lane & 15;
  const int hiA = (lane >> 4) * 8;
  const int hiB = (lane >> 4) * 16;

  const size_t bh = (size_t)(b * HEADS + h);
  const int qBase = qblk * 16;

  // ---- Phase 1: scores = q . k^T, streamed to LDS ----
  const _Float16* qrow = qh + (bh * SEQ + qBase + n15) * HD + hiB;
  v16h aq0 = load_a_frag(qrow, qrow + 8);         // d = 0..31
  v16h aq1 = load_a_frag(qrow + 32, qrow + 40);   // d = 32..63

  for (int kt = 0; kt < 32; ++kt) {
    int keyBase = wave * 512 + kt * 16;
    const _Float16* krow = kh + (bh * SEQ + keyBase + n15) * HD + hiB;
    v8f acc = {};
    v16h b0 = *(const v16h*)(krow);
    acc = __builtin_amdgcn_wmma_f32_16x16x32_f16(false, aq0, false, b0,
                                                 (short)0, acc, false, false);
    v16h b1 = *(const v16h*)(krow + 32);
    acc = __builtin_amdgcn_wmma_f32_16x16x32_f16(false, aq1, false, b1,
                                                 (short)0, acc, false, false);
    int col = keyBase + n15;
#pragma unroll
    for (int r = 0; r < 8; ++r) sc[(r + hiA) * SEQ + col] = acc[r];
  }
  __syncthreads();

  // ---- Phase 2: softmax stats (8 threads per row, 256 cols each) ----
  {
    int r = tid >> 3;
    int c0 = (tid & 7) * 256;
    float lm = -3.0e38f;
    for (int c = 0; c < 256; ++c) {
      float v = sc[r * SEQ + c0 + c];
      lm = v > lm ? v : lm;
    }
    red[r * 16 + (tid & 7)] = lm;
    __syncthreads();
    if (tid < 16) {
      float m = red[tid * 16];
      for (int i = 1; i < 8; ++i) {
        float v = red[tid * 16 + i];
        m = v > m ? v : m;
      }
      rowmax[tid] = m;
    }
    __syncthreads();
    float m = rowmax[r];
    float ls = 0.0f;
    for (int c = 0; c < 256; ++c) {
      float e = __expf(sc[r * SEQ + c0 + c] - m);
      p[r * SEQ + c0 + c] = (_Float16)e;   // unnormalized P
      ls += e;
    }
    red[r * 16 + (tid & 7)] = ls;
    __syncthreads();
    if (tid < 16) {
      float s = 0.0f;
      for (int i = 0; i < 8; ++i) s += red[tid * 16 + i];
      rowsum[tid] = s;
    }
    __syncthreads();
  }

  // ---- Phase 3: P * V over this wave's 512 keys ----
  v8f zero = {};
  v8f oacc[4];
#pragma unroll
  for (int j = 0; j < 4; ++j) oacc[j] = zero;

  const _Float16* vbase = vth + bh * HD * SEQ;
  for (int kt = 0; kt < 16; ++kt) {
    int kb = wave * 512 + kt * 32;
    const _Float16* prow = p + n15 * SEQ + kb + hiB;
    v16h a = load_a_frag(prow, prow + 8);
#pragma unroll
    for (int j = 0; j < 4; ++j) {
      int d = j * 16 + n15;
      v16h bv = *(const v16h*)(vbase + (size_t)d * SEQ + kb + hiB);
      oacc[j] = __builtin_amdgcn_wmma_f32_16x16x32_f16(
          false, a, false, bv, (short)0, oacc[j], false, false);
    }
  }

  // write per-wave partials (reuses sc region; all sc reads are done)
#pragma unroll
  for (int j = 0; j < 4; ++j) {
    int col = j * 16 + n15;
#pragma unroll
    for (int r = 0; r < 8; ++r)
      outred[(wave * 16 + r + hiA) * 64 + col] = oacc[j][r];
  }
  __syncthreads();

  // ---- Phase 4: reduce 4 waves, normalize, store f16 [b, s, h*64+d] ----
  for (int i = 0; i < 8; ++i) {
    int e = tid * 8 + i;
    int row = e >> 6;
    int col = e & 63;
    float s = outred[row * 64 + col] + outred[(16 + row) * 64 + col] +
              outred[(32 + row) * 64 + col] + outred[(48 + row) * 64 + col];
    float val = s / rowsum[row];
    attnh[((size_t)(b * SEQ + qBase + row)) * DIMS + h * HD + col] =
        (_Float16)val;
  }
}

// ---------------------------------------------------------------------------
extern "C" void kernel_launch(void* const* d_in, const int* in_sizes, int n_in,
                              void* d_out, int out_size, void* d_ws,
                              size_t ws_size, hipStream_t stream) {
  const float* x     = (const float*)d_in[0];  // [2,2048,1024]
  const float* w_qkv = (const float*)d_in[1];  // [3072,1024]
  const float* w_out = (const float*)d_in[2];  // [1024,1024]
  const float* b_out = (const float*)d_in[3];  // [1024]
  float* out = (float*)d_out;                  // [2,2048,1024]

  char* ws = (char*)d_ws;
  _Float16* xh    = (_Float16*)ws; ws += (size_t)MTOT * DIMS * 2;       // 8 MB
  _Float16* wqkvh = (_Float16*)ws; ws += (size_t)3 * DIMS * DIMS * 2;   // 6 MB
  _Float16* wouth = (_Float16*)ws; ws += (size_t)DIMS * DIMS * 2;       // 2 MB
  _Float16* qhb   = (_Float16*)ws; ws += (size_t)MTOT * DIMS * 2;       // 8 MB
  _Float16* khb   = (_Float16*)ws; ws += (size_t)MTOT * DIMS * 2;       // 8 MB
  _Float16* vtb   = (_Float16*)ws; ws += (size_t)MTOT * DIMS * 2;       // 8 MB
  _Float16* attnh = (_Float16*)ws; ws += (size_t)MTOT * DIMS * 2;       // 8 MB

  // 1. convert inputs to f16
  f32_to_f16_kernel<<<1024, 256, 0, stream>>>(x, xh, MTOT * DIMS);
  f32_to_f16_kernel<<<1024, 256, 0, stream>>>(w_qkv, wqkvh, 3 * DIMS * DIMS);
  f32_to_f16_kernel<<<512, 256, 0, stream>>>(w_out, wouth, DIMS * DIMS);

  // 2. QKV projection: [4096,1024] x [3072,1024]^T, routed to q/k/v^T
  gemm_f16_kernel<<<dim3(MTOT / 128, (3 * DIMS) / 64), 128, 0, stream>>>(
      xh, wqkvh, DIMS, 3 * DIMS, /*mode=*/0, qhb, khb, vtb, nullptr, nullptr);

  // 3. attention: (b, h, qblock) grid
  const size_t attn_lds = 131072 + 65536 + (16 * 16 + 16 + 16) * sizeof(float);
  attn_kernel<<<BATCH * HEADS * (SEQ / 16), 128, attn_lds, stream>>>(
      qhb, khb, vtb, attnh);

  // 4. output projection + bias: [4096,1024] x [1024,1024]^T -> fp32
  gemm_f16_kernel<<<dim3(MTOT / 128, DIMS / 64), 128, 0, stream>>>(
      attnh, wouth, DIMS, DIMS, /*mode=*/1, nullptr, nullptr, nullptr, out,
      b_out);
}
